// MPNN_70626442215972
// MI455X (gfx1250) — compile-verified, matrix operand order checked by reading, not run on hardware
//
#include <hip/hip_runtime.h>

// ---------------------------------------------------------------------------
// MPNN on gfx1250 (MI455X), wave32 + v_wmma_f32_16x16x32_bf16.
//
// Factorization:  M_w = [M_src; M_dst; M_e],  U_w = [U_h; U_n]
//   P  = bf16( h @ [M_src | M_dst] )          [N, 256]  (bf16 workspace)
//   HU = h @ U_h                              [N, 128]  (f32 workspace)
//   per edge: m = LeakyReLU(P[src,0:128] + P[dst,128:256] + e @ M_e)
//   s[dst] += m (f32 atomics), cnt[dst] += 1
//   out = LeakyReLU(HU + (s/max(cnt,1)) @ U_n)
//
// All weight B-fragments are pre-packed to bf16 (fragment-major) once per
// call so GEMM waves load them with coalesced b128 loads.
// ---------------------------------------------------------------------------

typedef __attribute__((ext_vector_type(16))) __bf16 v16bf;
typedef __attribute__((ext_vector_type(8)))  __bf16 v8bf;
typedef __attribute__((ext_vector_type(8)))  float  v8f;

#define IN_CH  64
#define OUT_CH 128
#define EDGE_F 32
#define SLOPE  0.01f

// Fragment table inside WPACK (each fragment = 32 lanes * 16 bf16 = 1 KB):
//   [0..7]   : M_e,  k0=0,        n0=f*16                  (edge kernel)
//   [8..55]  : node kernel, f=8+nt*2+kf, nt in 0..23, kf in 0..1
//   [56..87] : U_n,  f=56+nt*4+kf, nt in 0..7,  kf in 0..3 (update kernel)
#define FRAG_EDGE  0
#define FRAG_NODE  8
#define FRAG_UPD   56
#define N_FRAGS    88

static __device__ __forceinline__ int imin(int a, int b) { return a < b ? a : b; }

// Compiler-level memory barrier: keeps LDS store->load ordering (hardware DS
// ops from the same wave are processed in order).
static __device__ __forceinline__ void wave_mem_fence() {
  __builtin_amdgcn_wave_barrier();
  asm volatile("" ::: "memory");
}

// A-fragment (16x32 bf16, M x K) from row-major f32 source, float4 loads.
// ISA layout: lanes 0-15 -> row=lane, K {0..7,16..23};
//             lanes 16-31 -> row=lane-16, K {8..15,24..31}.
static __device__ __forceinline__ v16bf
a_frag_f32(const float* __restrict__ src, long row0, int stride, int kstart,
           int lane, int rows_left) {
  int r  = lane & 15;
  int kb = kstart + ((lane < 16) ? 0 : 8);
  long rr = row0 + imin(r, rows_left - 1);
  const float4* p = (const float4*)(src + rr * stride + kb);
  float4 x0 = p[0], x1 = p[1];   // K = kb .. kb+7
  float4 y0 = p[4], y1 = p[5];   // K = kb+16 .. kb+23
  v16bf a;
  a[0]  = (__bf16)x0.x; a[1]  = (__bf16)x0.y; a[2]  = (__bf16)x0.z; a[3]  = (__bf16)x0.w;
  a[4]  = (__bf16)x1.x; a[5]  = (__bf16)x1.y; a[6]  = (__bf16)x1.z; a[7]  = (__bf16)x1.w;
  a[8]  = (__bf16)y0.x; a[9]  = (__bf16)y0.y; a[10] = (__bf16)y0.z; a[11] = (__bf16)y0.w;
  a[12] = (__bf16)y1.x; a[13] = (__bf16)y1.y; a[14] = (__bf16)y1.z; a[15] = (__bf16)y1.w;
  return a;
}

// A-fragment from a bf16 LDS tile (row-major), v8bf loads.
static __device__ __forceinline__ v16bf
a_frag_lds(const __bf16* __restrict__ src, int stride, int kstart, int lane) {
  int r  = lane & 15;
  int kb = kstart + ((lane < 16) ? 0 : 8);
  const __bf16* p = src + r * stride + kb;
  v8bf lo = *(const v8bf*)p;          // K = kb .. kb+7
  v8bf hi = *(const v8bf*)(p + 16);   // K = kb+16 .. kb+23
  v16bf a;
#pragma unroll
  for (int j = 0; j < 8; ++j) { a[j] = lo[j]; a[8 + j] = hi[j]; }
  return a;
}

// B-fragment (32x16 bf16, K x N) from row-major f32 weights (pack kernel only).
static __device__ __forceinline__ v16bf
b_frag_f32(const float* __restrict__ W, int ldn, int k0, int n0, int lane) {
  int n  = n0 + (lane & 15);
  int kb = k0 + ((lane < 16) ? 0 : 8);
  const float* p = W + (long)kb * ldn + n;
  v16bf b;
#pragma unroll
  for (int j = 0; j < 8; ++j) b[j]     = (__bf16)p[(long)j * ldn];
#pragma unroll
  for (int j = 0; j < 8; ++j) b[8 + j] = (__bf16)p[(long)(16 + j) * ldn];
  return b;
}

static __device__ __forceinline__ v8f wmma_bf16(v16bf a, v16bf b, v8f c) {
  return __builtin_amdgcn_wmma_f32_16x16x32_bf16(
      false, a, false, b, (short)0, c, false, false);
}

static __device__ __forceinline__ float leaky(float v) {
  return (v > 0.0f) ? v : SLOPE * v;
}

// ---------------------------------------------------------------------------
// Kernel 0: pack all weight B-fragments to bf16, fragment-major.
// One wave per fragment (88 waves).
// ---------------------------------------------------------------------------
__global__ void pack_weights_kernel(const float* __restrict__ Mw,  // 160 x 128
                                    const float* __restrict__ Uw,  // 192 x 128
                                    v16bf* __restrict__ Wp) {
  int f    = (blockIdx.x * blockDim.x + threadIdx.x) >> 5;
  int lane = threadIdx.x & 31;
  if (f >= N_FRAGS) return;

  const float* W;
  int k0, n0;
  if (f < FRAG_NODE) {                       // M_e
    W = Mw + 128 * OUT_CH; k0 = 0; n0 = f * 16;
  } else if (f < FRAG_UPD) {                 // node kernel: [M_src|M_dst|U_h]
    int t = f - FRAG_NODE, nt = t >> 1, kf = t & 1;
    k0 = kf * 32;
    if (nt < 8)       { W = Mw;               n0 = nt * 16; }
    else if (nt < 16) { W = Mw + 64 * OUT_CH; n0 = (nt - 8) * 16; }
    else              { W = Uw;               n0 = (nt - 16) * 16; }
  } else {                                   // U_n
    int t = f - FRAG_UPD, nt = t >> 2, kf = t & 3;
    W = Uw + 64 * OUT_CH; k0 = kf * 32; n0 = nt * 16;
  }
  Wp[f * 32 + lane] = b_frag_f32(W, OUT_CH, k0, n0, lane);
}

// ---------------------------------------------------------------------------
// Kernel 1: P = bf16(h @ [M_src | M_dst]),  HU = h @ U_h
// One wave per 16-node tile; 24 column tiles, K = 64 (2 chained WMMAs).
// ---------------------------------------------------------------------------
__global__ void node_proj_kernel(const float* __restrict__ h,
                                 const v16bf* __restrict__ Wp,
                                 __bf16* __restrict__ P,     // N x 256
                                 float* __restrict__ HU,     // N x 128
                                 int n_nodes, int n_tiles) {
  int wave = (blockIdx.x * blockDim.x + threadIdx.x) >> 5;
  int lane = threadIdx.x & 31;
  if (wave >= n_tiles) return;
  long base = (long)wave * 16;
  bool full = (base + 16) <= n_nodes;

  v16bf a0 = a_frag_f32(h, base, IN_CH, 0,  lane, n_nodes - (int)base);
  v16bf a1 = a_frag_f32(h, base, IN_CH, 32, lane, n_nodes - (int)base);

  int col   = lane & 15;
  int rbase = (lane < 16) ? 0 : 8;

#pragma unroll 1
  for (int nt = 0; nt < 24; ++nt) {
    v16bf b0 = Wp[(FRAG_NODE + nt * 2 + 0) * 32 + lane];
    v16bf b1 = Wp[(FRAG_NODE + nt * 2 + 1) * 32 + lane];
    v8f c = {};
    c = wmma_bf16(a0, b0, c);
    c = wmma_bf16(a1, b1, c);

    if (nt < 16) {
      int pcol = nt * 16 + col;                // 0..255
      if (full) {
#pragma unroll
        for (int i = 0; i < 8; ++i)
          P[(base + rbase + i) * 256 + pcol] = (__bf16)c[i];
      } else {
#pragma unroll
        for (int i = 0; i < 8; ++i) {
          long r = base + rbase + i;
          if (r < n_nodes) P[r * 256 + pcol] = (__bf16)c[i];
        }
      }
    } else {
      int ucol = (nt - 16) * 16 + col;         // 0..127
      if (full) {
#pragma unroll
        for (int i = 0; i < 8; ++i)
          HU[(base + rbase + i) * OUT_CH + ucol] = c[i];
      } else {
#pragma unroll
        for (int i = 0; i < 8; ++i) {
          long r = base + rbase + i;
          if (r < n_nodes) HU[r * OUT_CH + ucol] = c[i];
        }
      }
    }
  }
}

// ---------------------------------------------------------------------------
// Kernel 2: per 16-edge tile:
//   m = LeakyReLU(e@M_e + P[src,:128] + P[dst,128:]); s[dst]+=m; cnt[dst]+=1.
// 128-thread blocks (4 waves); grid-stride over tiles; B fragments live in
// 64 VGPRs for the wave's whole lifetime. P rows staged via b128 into LDS.
// ---------------------------------------------------------------------------
#define EDGE_WAVES 4
__global__ void __launch_bounds__(EDGE_WAVES * 32)
edge_msg_kernel(const float* __restrict__ e,     // E x 32
                const int* __restrict__ src,
                const int* __restrict__ dst,
                const v16bf* __restrict__ Wp,
                const __bf16* __restrict__ P,    // N x 256
                float* __restrict__ s,           // N x 128
                float* __restrict__ cnt,         // N
                int n_edges, int n_tiles, int wstride) {
  __shared__ __bf16 gbuf[EDGE_WAVES][16 * 256];  // 4 x 8 KB
  __shared__ int    gdst[EDGE_WAVES][16];

  int wave_in_blk = threadIdx.x >> 5;
  int lane        = threadIdx.x & 31;
  int wave        = (blockIdx.x * blockDim.x + threadIdx.x) >> 5;

  // B fragments (M_e), resident for the whole wave.
  v16bf bfr[8];
#pragma unroll
  for (int nt = 0; nt < 8; ++nt) bfr[nt] = Wp[(FRAG_EDGE + nt) * 32 + lane];

  __bf16* gb = gbuf[wave_in_blk];
  int*    gd = gdst[wave_in_blk];
  int col    = lane & 15;
  int rbase  = (lane < 16) ? 0 : 8;
  int r      = lane & 15;
  int part   = lane >> 4;       // 0: gather P[src,:128], 1: gather P[dst,128:]

#pragma unroll 1
  for (int tile = wave; tile < n_tiles; tile += wstride) {
    long base = (long)tile * 16;
    bool full = (base + 16) <= n_edges;

    // Prefetch next tile's inputs while we work on this one.
    long nb = (long)(tile + wstride) * 16;
    if (nb < n_edges) {
      __builtin_prefetch(e + (nb + r) * EDGE_F, 0, 0);
      if (part == 0) {
        __builtin_prefetch(&src[nb + r], 0, 0);
        __builtin_prefetch(&dst[nb + r], 0, 0);
      }
    }

    // A fragment: 16 edges x 32 edge-features.
    v16bf a = a_frag_f32(e, base, EDGE_F, 0, lane, n_edges - (int)base);

    // Gather: lane r/part fetches one 256B half-row of P into LDS (b128s).
    long eidx = base + imin(r, n_edges - 1 - (int)base);
    int  node = part ? dst[eidx] : src[eidx];
    if (part) gd[r] = node;
    {
      const v8bf* prow = (const v8bf*)(P + (long)node * 256 + part * 128);
      v8bf*       drow = (v8bf*)(gb + r * 256 + part * 128);
#pragma unroll
      for (int j = 0; j < 16; ++j) drow[j] = prow[j];
    }
    wave_mem_fence();   // LDS ops from one wave are processed in order

#pragma unroll
    for (int nt = 0; nt < 8; ++nt) {
      int n0 = nt * 16;
      v8f c = {};
      c = wmma_bf16(a, bfr[nt], c);
      if (full) {
#pragma unroll
        for (int i = 0; i < 8; ++i) {
          const __bf16* row = gb + (rbase + i) * 256 + n0 + col;
          float v = leaky(c[i] + (float)row[0] + (float)row[128]);
          atomicAdd(&s[(long)gd[rbase + i] * OUT_CH + (n0 + col)], v);
        }
      } else {
#pragma unroll
        for (int i = 0; i < 8; ++i) {
          if (base + rbase + i < n_edges) {
            const __bf16* row = gb + (rbase + i) * 256 + n0 + col;
            float v = leaky(c[i] + (float)row[0] + (float)row[128]);
            atomicAdd(&s[(long)gd[rbase + i] * OUT_CH + (n0 + col)], v);
          }
        }
      }
    }

    if (lane < 16 && (full || base + lane < n_edges))
      atomicAdd(&cnt[gd[lane]], 1.0f);
    wave_mem_fence();   // don't let next tile's gather overtake the reads
  }
}

// ---------------------------------------------------------------------------
// Kernel 3: out = LeakyReLU(HU + (s / max(cnt,1)) @ U_n).
// One wave per 16-node tile; h_N staged in LDS as bf16; K = 128 (4 WMMAs).
// ---------------------------------------------------------------------------
__global__ void update_kernel(const float* __restrict__ s,     // N x 128
                              const float* __restrict__ cnt,   // N
                              const v16bf* __restrict__ Wp,
                              const float* __restrict__ HU,    // N x 128
                              float* __restrict__ out,         // N x 128
                              int n_nodes, int n_tiles) {
  __shared__ __bf16 lds[8][16 * OUT_CH];  // 8 waves * 4 KB = 32 KB
  int wave_in_blk = threadIdx.x >> 5;
  int lane        = threadIdx.x & 31;
  int wave        = (blockIdx.x * blockDim.x + threadIdx.x) >> 5;
  bool active     = wave < n_tiles;
  long base       = active ? (long)wave * 16 : 0;
  bool full       = active && (base + 16) <= n_nodes;

  __bf16* hn = lds[wave_in_blk];
  if (active) {
    for (int idx = lane; idx < 16 * OUT_CH; idx += 32) {
      int rr = idx >> 7, cc = idx & 127;
      long r = base + imin(rr, n_nodes - 1 - (int)base);
      float v = s[r * OUT_CH + cc] / fmaxf(cnt[r], 1.0f);
      hn[idx] = (__bf16)v;
    }
  }
  __syncthreads();
  if (!active) return;

  v16bf a0 = a_frag_lds(hn, OUT_CH, 0,  lane);
  v16bf a1 = a_frag_lds(hn, OUT_CH, 32, lane);
  v16bf a2 = a_frag_lds(hn, OUT_CH, 64, lane);
  v16bf a3 = a_frag_lds(hn, OUT_CH, 96, lane);

  int col   = lane & 15;
  int rbase = (lane < 16) ? 0 : 8;

#pragma unroll 1
  for (int nt = 0; nt < 8; ++nt) {
    int n0 = nt * 16;
    v8f c;
#pragma unroll
    for (int i = 0; i < 8; ++i) {
      long r = base + imin(rbase + i, n_nodes - 1 - (int)base);
      c[i] = HU[r * OUT_CH + (n0 + col)];
    }
    c = wmma_bf16(a0, Wp[(FRAG_UPD + nt * 4 + 0) * 32 + lane], c);
    c = wmma_bf16(a1, Wp[(FRAG_UPD + nt * 4 + 1) * 32 + lane], c);
    c = wmma_bf16(a2, Wp[(FRAG_UPD + nt * 4 + 2) * 32 + lane], c);
    c = wmma_bf16(a3, Wp[(FRAG_UPD + nt * 4 + 3) * 32 + lane], c);

    if (full) {
#pragma unroll
      for (int i = 0; i < 8; ++i)
        out[(base + rbase + i) * OUT_CH + (n0 + col)] = leaky(c[i]);
    } else {
#pragma unroll
      for (int i = 0; i < 8; ++i) {
        long r = base + rbase + i;
        if (r < n_nodes) out[r * OUT_CH + (n0 + col)] = leaky(c[i]);
      }
    }
  }
}

// ---------------------------------------------------------------------------
extern "C" void kernel_launch(void* const* d_in, const int* in_sizes, int n_in,
                              void* d_out, int out_size, void* d_ws, size_t ws_size,
                              hipStream_t stream) {
  const float* h   = (const float*)d_in[0];
  const float* e   = (const float*)d_in[1];
  const int*   src = (const int*)d_in[2];
  const int*   dst = (const int*)d_in[3];
  const float* Mw  = (const float*)d_in[4];
  const float* Uw  = (const float*)d_in[5];
  float*       out = (float*)d_out;

  int n_nodes = in_sizes[0] / IN_CH;
  int n_edges = in_sizes[2];

  char* ws = (char*)d_ws;
  size_t szW  = (size_t)N_FRAGS * 32 * sizeof(v16bf);       // 88 KB, 256B-aligned
  size_t szP  = (size_t)n_nodes * 256 * sizeof(__bf16);
  size_t szHU = (size_t)n_nodes * OUT_CH * sizeof(float);
  size_t szS  = (size_t)n_nodes * OUT_CH * sizeof(float);
  size_t szC  = (size_t)n_nodes * sizeof(float);

  v16bf*  Wp = (v16bf*)ws;
  __bf16* P  = (__bf16*)(ws + szW);
  float*  HU = (float*)(ws + szW + szP);
  float*  S  = (float*)(ws + szW + szP + szHU);
  float*  C  = (float*)(ws + szW + szP + szHU + szS);

  // Zero the accumulators every call (S and C are contiguous).
  hipMemsetAsync(S, 0, szS + szC, stream);

  int tilesN = (n_nodes + 15) / 16;
  int tilesE = (n_edges + 15) / 16;

  pack_weights_kernel<<<(N_FRAGS * 32 + 255) / 256, 256, 0, stream>>>(Mw, Uw, Wp);

  node_proj_kernel<<<(tilesN + 7) / 8, 256, 0, stream>>>(
      h, Wp, P, HU, n_nodes, tilesN);

  // ~4 tiles per wave, 4 waves per block.
  int ewaves  = (tilesE + 3) / 4;
  int eblocks = (ewaves + EDGE_WAVES - 1) / EDGE_WAVES;
  edge_msg_kernel<<<eblocks, EDGE_WAVES * 32, 0, stream>>>(
      e, src, dst, Wp, P, S, C, n_edges, tilesE, eblocks * EDGE_WAVES);

  update_kernel<<<(tilesN + 7) / 8, 256, 0, stream>>>(
      S, C, Wp, HU, out, n_nodes, tilesN);
}
